// InternAttention_15642270892158
// MI455X (gfx1250) — matrix operand order repaired and checked by reading, not compiled, not run on hardware
//
#include <hip/hip_runtime.h>
#include <hip/hip_bf16.h>

typedef __bf16 bf16;
typedef __attribute__((ext_vector_type(16))) __bf16 v16bf;
typedef __attribute__((ext_vector_type(8)))  __bf16 v8bf;
typedef __attribute__((ext_vector_type(4)))  __bf16 v4bf;
typedef __attribute__((ext_vector_type(8)))  float  v8f;
typedef __attribute__((ext_vector_type(4)))  float  v4f;

#define DEVINL __device__ __forceinline__

// ---------------- WMMA fragment helpers (CDNA5 16x16x32 bf16) ----------------

DEVINL v16bf cat16(v8bf lo, v8bf hi) {
  return __builtin_shufflevector(lo, hi, 0,1,2,3,4,5,6,7,8,9,10,11,12,13,14,15);
}

DEVINL v8f wmma_bf16(v16bf a, v16bf b, v8f c) {
  // D = A(16x32) * B(32x16) + C(16x16 f32)
  return __builtin_amdgcn_wmma_f32_16x16x32_bf16(
      /*neg_a=*/false, a, /*neg_b=*/false, b,
      /*c_mod=*/(short)0, c, /*reuse_a=*/false, /*reuse_b=*/false);
}

// A-operand: 16x32 row-major (row stride ld elements given by caller pointer math).
// lane<16: row=lane, K {kk..kk+7, kk+16..kk+23}; lane>=16: K {kk+8.., kk+24..}.
DEVINL v16bf load_a_frag(const bf16* __restrict__ row, int kk, int half) {
  v8bf lo = *(const v8bf*)(row + kk + half * 8);
  v8bf hi = *(const v8bf*)(row + kk + 16 + half * 8);
  return cat16(lo, hi);
}

// B-operand: 32x16 with B(k,n) = M[n][k] for row-major M.
// lane<16: col n=lane, K {kk..kk+7, kk+16..kk+23}; lane>=16: col n=lane-16, K {kk+8.., kk+24..}.
DEVINL v16bf load_b_frag(const bf16* __restrict__ row, int kk, int half) {
  v8bf lo = *(const v8bf*)(row + kk + half * 8);
  v8bf hi = *(const v8bf*)(row + kk + 16 + half * 8);
  return cat16(lo, hi);
}

// ---------------- CDNA5 async global->LDS (ASYNCcnt path) ----------------

DEVINL unsigned lds_off_of(const void* p) {
  // Generic (flat) LDS pointer: low 32 bits are the LDS byte address (ISA 10.2).
  return (unsigned)(unsigned long long)p;
}

DEVINL void async_ld_b128(void* lds_dst, const void* gsrc) {
  unsigned l = lds_off_of(lds_dst);
  asm volatile("global_load_async_to_lds_b128 %0, %1, off"
               :: "v"(l), "v"(gsrc) : "memory");
}

DEVINL void wait_async0()   { asm volatile("s_wait_asynccnt 0x0" ::: "memory"); }
DEVINL void wait_async_le2(){ asm volatile("s_wait_asynccnt 0x2" ::: "memory"); }
DEVINL void wait_async_le4(){ asm volatile("s_wait_asynccnt 0x4" ::: "memory"); }

// ---------------- Kernel 0: fp32 -> bf16 conversion ----------------

__global__ void __launch_bounds__(256)
cvt_f32_bf16_kernel(const float* __restrict__ in, bf16* __restrict__ out, int n) {
  int idx = (blockIdx.x * 256 + threadIdx.x) * 4;
  if (idx + 3 < n) {
    v4f f = *(const v4f*)(in + idx);
    v4bf o;
    o[0] = (bf16)f[0]; o[1] = (bf16)f[1]; o[2] = (bf16)f[2]; o[3] = (bf16)f[3];
    *(v4bf*)(out + idx) = o;
  }
}

// ---------------- QKV epilogue ----------------
// q,k stored fp32 token-major; v stored bf16 transposed [B,H,D,N].

DEVINL void qkv_store_tile(v8f acc, int n, int row0, const float* __restrict__ bias,
                           float* __restrict__ qraw, float* __restrict__ kraw,
                           bf16* __restrict__ vt) {
  const float bv = bias[n];
  if (n < 1024) {                               // Q (fp32 for full-width RMSNorm)
#pragma unroll
    for (int i = 0; i < 8; i++)
      qraw[(size_t)(row0 + i) * 1024 + n] = acc[i] + bv;
  } else if (n < 2048) {                        // K
    const int f = n - 1024;
#pragma unroll
    for (int i = 0; i < 8; i++)
      kraw[(size_t)(row0 + i) * 1024 + f] = acc[i] + bv;
  } else {                                      // V -> bf16, transposed [B,H,D,N]
    const int f = n - 2048, h = f >> 6, d = f & 63;
    const int bidx = row0 >> 10, tok = row0 & 1023;
    v8bf pk;
#pragma unroll
    for (int i = 0; i < 8; i++) pk[i] = (bf16)(acc[i] + bv);
    *(v8bf*)(vt + (((size_t)(bidx * 16 + h) * 64 + d) * 1024 + tok)) = pk;
  }
}

// ---------------- Kernel 1: QKV GEMM (double-buffered async LDS pipeline) ----------------
// Block = 8 waves computes a 64x64 tile of x[16384,1024] * W[3072,1024]^T.
// K-slab = 64: A/B stage tiles 64x64 bf16 (8 KB each), double buffered.
// Each thread stages 4x16B chunks per slab; steady-state wait is asynccnt<=4.

__global__ void __launch_bounds__(256)
qkv_gemm_kernel(const bf16* __restrict__ xb, const bf16* __restrict__ wb,
                const float* __restrict__ bias,
                float* __restrict__ qraw, float* __restrict__ kraw,
                bf16* __restrict__ vt) {
  __shared__ __align__(16) bf16 As[2][64 * 64];   // 16 KB
  __shared__ __align__(16) bf16 Bs[2][64 * 64];   // 16 KB
  const int t = threadIdx.x;
  const int lane = t & 31, wave = t >> 5;
  const int half = lane >> 4, l15 = lane & 15;
  const int msub = wave & 3, npair = (wave >> 2) * 2;
  const int mblock = blockIdx.x * 64, nblock = blockIdx.y * 64;

  // Staging chunks: row r0 = t>>3 (0..31) and r0+32; col chunk c8 = (t&7)*8.
  const int r0 = t >> 3, c8 = (t & 7) * 8;
  const bf16* ag0 = xb + (size_t)(mblock + r0) * 1024 + c8;
  const bf16* ag1 = ag0 + (size_t)32 * 1024;
  const bf16* bg0 = wb + (size_t)(nblock + r0) * 1024 + c8;
  const bf16* bg1 = bg0 + (size_t)32 * 1024;

  v8f c0 = {0.f,0.f,0.f,0.f,0.f,0.f,0.f,0.f}, c1 = c0;

  // Prologue: stage slab 0 into buffer 0.
  async_ld_b128(&As[0][r0 * 64 + c8],        ag0);
  async_ld_b128(&As[0][(r0 + 32) * 64 + c8], ag1);
  async_ld_b128(&Bs[0][r0 * 64 + c8],        bg0);
  async_ld_b128(&Bs[0][(r0 + 32) * 64 + c8], bg1);

  for (int k0 = 0; k0 < 1024; k0 += 64) {
    const int buf = (k0 >> 6) & 1;
    if (k0 + 64 < 1024) {                       // stage next slab into other buffer
      const int nb = buf ^ 1;
      async_ld_b128(&As[nb][r0 * 64 + c8],        ag0 + k0 + 64);
      async_ld_b128(&As[nb][(r0 + 32) * 64 + c8], ag1 + k0 + 64);
      async_ld_b128(&Bs[nb][r0 * 64 + c8],        bg0 + k0 + 64);
      async_ld_b128(&Bs[nb][(r0 + 32) * 64 + c8], bg1 + k0 + 64);
      wait_async_le4();                         // only the previous slab must be done
    } else {
      wait_async0();
    }
    __syncthreads();

    const bf16* arow  = &As[buf][(msub * 16 + l15) * 64];
    const bf16* brow0 = &Bs[buf][(npair * 16 + l15) * 64];
    const bf16* brow1 = brow0 + 16 * 64;
#pragma unroll
    for (int kk = 0; kk < 64; kk += 32) {
      v16bf a = load_a_frag(arow, kk, half);
      c0 = wmma_bf16(a, load_b_frag(brow0, kk, half), c0);
      c1 = wmma_bf16(a, load_b_frag(brow1, kk, half), c1);
    }
    __syncthreads();                            // all reads done before re-staging buf
  }

  const int row0 = mblock + msub * 16 + half * 8;
  const int n0 = nblock + npair * 16 + l15;
  qkv_store_tile(c0, n0,      row0, bias, qraw, kraw, vt);
  qkv_store_tile(c1, n0 + 16, row0, bias, qraw, kraw, vt);
}

// ---------------- Kernel 2: full-width RMSNorm on q,k ----------------

__global__ void __launch_bounds__(256)
qk_norm_kernel(const float* __restrict__ qraw, const float* __restrict__ kraw,
               const float* __restrict__ qw, const float* __restrict__ kw,
               bf16* __restrict__ qn, bf16* __restrict__ kn) {
  const int m = blockIdx.x;                     // global token 0..16383
  const int t = threadIdx.x;
  const float* qp = qraw + (size_t)m * 1024;
  const float* kp = kraw + (size_t)m * 1024;

  float qv[4], kv[4], qs = 0.f, ks = 0.f;
#pragma unroll
  for (int j = 0; j < 4; j++) {
    qv[j] = qp[t + 256 * j]; kv[j] = kp[t + 256 * j];
    qs += qv[j] * qv[j];     ks += kv[j] * kv[j];
  }
#pragma unroll
  for (int off = 16; off; off >>= 1) {
    qs += __shfl_xor(qs, off, 32);
    ks += __shfl_xor(ks, off, 32);
  }
  __shared__ float sq[8], sk[8];
  if ((t & 31) == 0) { sq[t >> 5] = qs; sk[t >> 5] = ks; }
  __syncthreads();
  float aq = 0.f, ak = 0.f;
#pragma unroll
  for (int i = 0; i < 8; i++) { aq += sq[i]; ak += sk[i]; }

  const float rq = rsqrtf(aq * (1.0f / 1024.0f) + 1e-6f) * 0.125f; // * D^-0.5
  const float rk = rsqrtf(ak * (1.0f / 1024.0f) + 1e-6f);

  const int bidx = m >> 10, tok = m & 1023;
#pragma unroll
  for (int j = 0; j < 4; j++) {
    const int f = t + 256 * j, h = f >> 6, d = f & 63;
    const size_t o = ((size_t)(bidx * 16 + h) * 1024 + tok) * 64 + d;
    qn[o] = (bf16)(qv[j] * rq * qw[f]);
    kn[o] = (bf16)(kv[j] * rk * kw[f]);
  }
}

// ---------------- Kernel 3: flash attention (double-buffered K/V staging) ----------------
// Block = one (b,h), 8 waves = 8 q-tiles of 16 rows. Per 32-key step the K tile
// (32x64) and V^T tile (64x32) are async-staged once and shared by all 8 waves;
// next step's tiles prefetch while the current step computes.

__global__ void __launch_bounds__(256)
flash_attn_kernel(const bf16* __restrict__ qn, const bf16* __restrict__ kn,
                  const bf16* __restrict__ vt, bf16* __restrict__ attn) {
  __shared__ __align__(16) bf16 Ks[2][32 * 64];   // 8 KB: [key_local][d]
  __shared__ __align__(16) bf16 Vs[2][64 * 32];   // 8 KB: [d][key_local]
  __shared__ __align__(16) bf16 psh[8][16 * 32];  // 8 KB: per-wave P tile
  const int t = threadIdx.x;
  const int lane = t & 31, wave = t >> 5;
  const int half = lane >> 4, l15 = lane & 15;
  const int bh = blockIdx.x;                    // b*16 + h
  const int qt = blockIdx.y * 8 + wave;         // q tile 0..63
  bf16* pw = &psh[wave][0];

  const bf16* Qb = qn + (size_t)bh * 1024 * 64;
  const bf16* Kb = kn + (size_t)bh * 1024 * 64;
  const bf16* Vb = vt + (size_t)bh * 64 * 1024;

  // Staging chunks: K: row=t>>3 (0..31), chunk=(t&7)*8; V: row d=t>>2, chunk=(t&3)*8.
  const bf16* kglob = Kb + (size_t)(t >> 3) * 64 + (t & 7) * 8;
  const unsigned koff = (t >> 3) * 64 + (t & 7) * 8;
  const bf16* vglob = Vb + (size_t)(t >> 2) * 1024 + (t & 3) * 8;
  const unsigned voff = (t >> 2) * 32 + (t & 3) * 8;

  const bf16* qrow = Qb + (size_t)(qt * 16 + l15) * 64;
  const v16bf qa0 = load_a_frag(qrow, 0, half);   // head-dim 0..31
  const v16bf qa1 = load_a_frag(qrow, 32, half);  // head-dim 32..63

  v8f o0 = {0.f,0.f,0.f,0.f,0.f,0.f,0.f,0.f}, o1 = o0, o2 = o0, o3 = o0;
  float mrow[8], lrow[8];
#pragma unroll
  for (int i = 0; i < 8; i++) { mrow[i] = -3.0e38f; lrow[i] = 0.f; }

  // Prologue: stage key-step 0 into buffer 0.
  async_ld_b128(&Ks[0][koff], kglob);
  async_ld_b128(&Vs[0][voff], vglob);

  for (int j = 0; j < 1024; j += 32) {
    const int buf = (j >> 5) & 1;
    if (j + 32 < 1024) {
      const int nb = buf ^ 1;
      async_ld_b128(&Ks[nb][koff], kglob + (size_t)(j + 32) * 64);
      async_ld_b128(&Vs[nb][voff], vglob + (j + 32));
      wait_async_le2();
    } else {
      wait_async0();
    }
    __syncthreads();

    const bf16* krow0 = &Ks[buf][l15 * 64];
    const bf16* krow1 = &Ks[buf][(16 + l15) * 64];
    v8f s0 = {0.f,0.f,0.f,0.f,0.f,0.f,0.f,0.f}, s1 = s0;
    s0 = wmma_bf16(qa0, load_b_frag(krow0, 0,  half), s0);
    s0 = wmma_bf16(qa1, load_b_frag(krow0, 32, half), s0);
    s1 = wmma_bf16(qa0, load_b_frag(krow1, 0,  half), s1);
    s1 = wmma_bf16(qa1, load_b_frag(krow1, 32, half), s1);

    float al[8];
#pragma unroll
    for (int i = 0; i < 8; i++) {
      float mx = fmaxf(s0[i], s1[i]);               // row-max over 32 keys
      mx = fmaxf(mx, __shfl_xor(mx, 1, 32));
      mx = fmaxf(mx, __shfl_xor(mx, 2, 32));
      mx = fmaxf(mx, __shfl_xor(mx, 4, 32));
      mx = fmaxf(mx, __shfl_xor(mx, 8, 32));
      const float mn = fmaxf(mrow[i], mx);
      const float a  = __expf(mrow[i] - mn);
      const float p0 = __expf(s0[i] - mn);
      const float p1 = __expf(s1[i] - mn);
      float rs = p0 + p1;                           // row-sum
      rs += __shfl_xor(rs, 1, 32);
      rs += __shfl_xor(rs, 2, 32);
      rs += __shfl_xor(rs, 4, 32);
      rs += __shfl_xor(rs, 8, 32);
      lrow[i] = lrow[i] * a + rs;
      mrow[i] = mn;
      al[i] = a;
      s0[i] = p0; s1[i] = p1;
    }
#pragma unroll
    for (int i = 0; i < 8; i++) {
      o0[i] *= al[i]; o1[i] *= al[i]; o2[i] *= al[i]; o3[i] *= al[i];
    }

    // P: C-layout -> per-wave LDS tile -> A-layout fragment (same-wave DS in order)
#pragma unroll
    for (int i = 0; i < 8; i++) {
      const int r = i + half * 8;
      pw[r * 32 + l15]      = (bf16)s0[i];
      pw[r * 32 + 16 + l15] = (bf16)s1[i];
    }
    asm volatile("s_wait_dscnt 0x0" ::: "memory");
    const v16bf pa = cat16(*(const v8bf*)(pw + l15 * 32 + half * 8),
                           *(const v8bf*)(pw + l15 * 32 + 16 + half * 8));

    const bf16* vb = &Vs[buf][l15 * 32 + half * 8]; // B-operand from staged V^T
    o0 = wmma_bf16(pa, cat16(*(const v8bf*)(vb),            *(const v8bf*)(vb + 16)),            o0);
    o1 = wmma_bf16(pa, cat16(*(const v8bf*)(vb + 16 * 32),  *(const v8bf*)(vb + 16 * 32 + 16)),  o1);
    o2 = wmma_bf16(pa, cat16(*(const v8bf*)(vb + 32 * 32),  *(const v8bf*)(vb + 32 * 32 + 16)),  o2);
    o3 = wmma_bf16(pa, cat16(*(const v8bf*)(vb + 48 * 32),  *(const v8bf*)(vb + 48 * 32 + 16)),  o3);
    __syncthreads();
  }

  const int bidx = bh >> 4, h = bh & 15;
  const int tok0 = qt * 16 + half * 8;
#pragma unroll
  for (int i = 0; i < 8; i++) {
    const float inv = 1.f / lrow[i];
    const size_t ro = ((size_t)bidx * 1024 + tok0 + i) * 1024 + h * 64 + l15;
    attn[ro]      = (bf16)(o0[i] * inv);
    attn[ro + 16] = (bf16)(o1[i] * inv);
    attn[ro + 32] = (bf16)(o2[i] * inv);
    attn[ro + 48] = (bf16)(o3[i] * inv);
  }
}

// ---------------- Kernel 4: output projection (same double-buffered scheme) ----------------

__global__ void __launch_bounds__(256)
proj_gemm_kernel(const bf16* __restrict__ attn, const bf16* __restrict__ pwb,
                 const float* __restrict__ bias, float* __restrict__ out) {
  __shared__ __align__(16) bf16 As[2][64 * 64];
  __shared__ __align__(16) bf16 Bs[2][64 * 64];
  const int t = threadIdx.x;
  const int lane = t & 31, wave = t >> 5;
  const int half = lane >> 4, l15 = lane & 15;
  const int msub = wave & 3, npair = (wave >> 2) * 2;
  const int mblock = blockIdx.x * 64, nblock = blockIdx.y * 64;

  const int r0 = t >> 3, c8 = (t & 7) * 8;
  const bf16* ag0 = attn + (size_t)(mblock + r0) * 1024 + c8;
  const bf16* ag1 = ag0 + (size_t)32 * 1024;
  const bf16* bg0 = pwb + (size_t)(nblock + r0) * 1024 + c8;
  const bf16* bg1 = bg0 + (size_t)32 * 1024;

  v8f c0 = {0.f,0.f,0.f,0.f,0.f,0.f,0.f,0.f}, c1 = c0;

  async_ld_b128(&As[0][r0 * 64 + c8],        ag0);
  async_ld_b128(&As[0][(r0 + 32) * 64 + c8], ag1);
  async_ld_b128(&Bs[0][r0 * 64 + c8],        bg0);
  async_ld_b128(&Bs[0][(r0 + 32) * 64 + c8], bg1);

  for (int k0 = 0; k0 < 1024; k0 += 64) {
    const int buf = (k0 >> 6) & 1;
    if (k0 + 64 < 1024) {
      const int nb = buf ^ 1;
      async_ld_b128(&As[nb][r0 * 64 + c8],        ag0 + k0 + 64);
      async_ld_b128(&As[nb][(r0 + 32) * 64 + c8], ag1 + k0 + 64);
      async_ld_b128(&Bs[nb][r0 * 64 + c8],        bg0 + k0 + 64);
      async_ld_b128(&Bs[nb][(r0 + 32) * 64 + c8], bg1 + k0 + 64);
      wait_async_le4();
    } else {
      wait_async0();
    }
    __syncthreads();

    const bf16* arow  = &As[buf][(msub * 16 + l15) * 64];
    const bf16* brow0 = &Bs[buf][(npair * 16 + l15) * 64];
    const bf16* brow1 = brow0 + 16 * 64;
#pragma unroll
    for (int kk = 0; kk < 64; kk += 32) {
      v16bf a = load_a_frag(arow, kk, half);
      c0 = wmma_bf16(a, load_b_frag(brow0, kk, half), c0);
      c1 = wmma_bf16(a, load_b_frag(brow1, kk, half), c1);
    }
    __syncthreads();
  }

  const int row0 = mblock + msub * 16 + half * 8;
  const int n0 = nblock + npair * 16 + l15;
  const float bv0 = bias[n0], bv1 = bias[n0 + 16];
#pragma unroll
  for (int i = 0; i < 8; i++) {
    out[(size_t)(row0 + i) * 1024 + n0]      = c0[i] + bv0;
    out[(size_t)(row0 + i) * 1024 + n0 + 16] = c1[i] + bv1;
  }
}

// ---------------- Host launch ----------------

extern "C" void kernel_launch(void* const* d_in, const int* in_sizes, int n_in,
                              void* d_out, int out_size, void* d_ws, size_t ws_size,
                              hipStream_t stream) {
  const float* x      = (const float*)d_in[0];   // [16,1024,1024]
  const float* qkv_w  = (const float*)d_in[1];   // [3072,1024]
  const float* qkv_b  = (const float*)d_in[2];   // [3072]
  const float* qnormw = (const float*)d_in[3];   // [1024]
  const float* knormw = (const float*)d_in[4];   // [1024]
  const float* proj_w = (const float*)d_in[5];   // [1024,1024]
  const float* proj_b = (const float*)d_in[6];   // [1024]
  float* out = (float*)d_out;                    // [16,1024,1024]

  char* ws = (char*)d_ws;
  size_t off = 0;
  auto alloc = [&](size_t bytes) -> void* {
    void* p = ws + off;
    off += (bytes + 255) & ~(size_t)255;
    return p;
  };
  bf16*  xb    = (bf16*)alloc((size_t)16384 * 1024 * 2);  // x bf16
  bf16*  wb    = (bf16*)alloc((size_t)3072 * 1024 * 2);   // qkv_w bf16
  bf16*  pwb   = (bf16*)alloc((size_t)1024 * 1024 * 2);   // proj_w bf16
  float* qraw  = (float*)alloc((size_t)16384 * 1024 * 4); // q pre-norm fp32
  float* kraw  = (float*)alloc((size_t)16384 * 1024 * 4); // k pre-norm fp32
  bf16*  qnb   = (bf16*)alloc((size_t)16384 * 1024 * 2);  // q normed [B,H,N,D]
  bf16*  knb   = (bf16*)alloc((size_t)16384 * 1024 * 2);  // k normed [B,H,N,D]
  bf16*  vtb   = (bf16*)alloc((size_t)16384 * 1024 * 2);  // v transposed [B,H,D,N]
  bf16*  attnb = (bf16*)alloc((size_t)16384 * 1024 * 2);  // attention out [B*N,C]
  (void)ws_size; (void)in_sizes; (void)n_in; (void)out_size;

  cvt_f32_bf16_kernel<<<16384, 256, 0, stream>>>(x, xb, 16384 * 1024);
  cvt_f32_bf16_kernel<<<3072, 256, 0, stream>>>(qkv_w, wb, 3072 * 1024);
  cvt_f32_bf16_kernel<<<1024, 256, 0, stream>>>(proj_w, pwb, 1024 * 1024);

  qkv_gemm_kernel<<<dim3(256, 48), 256, 0, stream>>>(xb, wb, qkv_b, qraw, kraw, vtb);
  qk_norm_kernel<<<16384, 256, 0, stream>>>(qraw, kraw, qnormw, knormw, qnb, knb);
  flash_attn_kernel<<<dim3(256, 8), 256, 0, stream>>>(qnb, knb, vtb, attnb);
  proj_gemm_kernel<<<dim3(256, 16), 256, 0, stream>>>(attnb, pwb, proj_b, out);
}
